// MultiHeadCrossAttention_87351044866307
// MI455X (gfx1250) — compile-verified, matrix operand order checked by reading, not run on hardware
//
#include <hip/hip_runtime.h>

typedef __bf16 bf16;
typedef __attribute__((ext_vector_type(16))) __bf16 v16bf;
typedef __attribute__((ext_vector_type(8)))  __bf16 v8bf;
typedef __attribute__((ext_vector_type(4)))  __bf16 v4bf;
typedef __attribute__((ext_vector_type(8)))  float  v8f;

#define BATCH      16384
#define TEXT_DIM   1024
#define IMAGE_DIM  2048
#define NUM_HEADS  8
#define HEAD_DIM   128

// ---------------------------------------------------------------------------
// Phase 1: f32 -> bf16 conversion (vectorized)
// ---------------------------------------------------------------------------
__global__ __launch_bounds__(256)
void f32_to_bf16_kernel(const float* __restrict__ in, bf16* __restrict__ out, int n) {
    int i = (blockIdx.x * blockDim.x + threadIdx.x) * 4;
    if (i + 3 < n) {
        float4 v = *reinterpret_cast<const float4*>(in + i);
        v4bf o;
        o[0] = (bf16)v.x; o[1] = (bf16)v.y; o[2] = (bf16)v.z; o[3] = (bf16)v.w;
        *reinterpret_cast<v4bf*>(out + i) = o;
    } else {
        for (; i < n; ++i) out[i] = (bf16)in[i];
    }
}

// ---------------------------------------------------------------------------
// Phase 2: bf16 WMMA GEMM: D[m,n] = sum_k A[m,k] * W[n,k] + bias[n]
//   A: [M, K] bf16 row-major, W: [N, K] bf16 row-major (torch layout).
// Per-lane WMMA 16-bit operand layout (ISA 7.12.2): lane L(0..15) = row L,
// holds K = {0..7, 16..23}; lanes 16..31: same rows, K = {8..15, 24..31}.
// A and B fragments load identically since both are K-contiguous.
//
// Wave tile 64x64 = 16 accumulators; 4 A-frags + 4 B-frags per K=32 step
// -> 16 v_wmma per 16 b128 loads. Wave-uniform bases forced to SGPRs so the
// K-loop advances with SALU only; the per-lane offset VGPR is loop-invariant.
// ---------------------------------------------------------------------------
__device__ __forceinline__ v16bf combine_frag(v8bf lo, v8bf hi) {
    return __builtin_shufflevector(lo, hi, 0, 1, 2, 3, 4, 5, 6, 7,
                                           8, 9, 10, 11, 12, 13, 14, 15);
}

__global__ __launch_bounds__(256)
void gemm_bf16_wmma(const bf16* __restrict__ A, const bf16* __restrict__ W,
                    const float* __restrict__ bias, bf16* __restrict__ D,
                    int Kdim) {
    const int lane  = threadIdx.x & 31;
    const int wave  = __builtin_amdgcn_readfirstlane(threadIdx.x >> 5);
    const int waveM = wave & 3;           // 4 waves along M
    const int waveN = wave >> 2;          // 2 waves along N
    const int m0 = blockIdx.y * 256 + waveM * 64;   // wave tile: 64 (M) x 64 (N)
    const int n0 = blockIdx.x * 128 + waveN * 64;

    // wave-uniform (scalar) base pointers, advanced by 32 elements per K-step
    const bf16* pA = A + (size_t)m0 * Kdim;
    const bf16* pW = W + (size_t)n0 * Kdim;

    // loop-invariant per-lane fragment offset (elements)
    const int fragOff = (lane & 15) * Kdim + ((lane >> 4) << 3);
    const int rowStride16 = 16 * Kdim;    // uniform -> folds into imm offset

    v8f acc[4][4];
#pragma unroll
    for (int i = 0; i < 4; ++i)
#pragma unroll
        for (int j = 0; j < 4; ++j)
#pragma unroll
            for (int v = 0; v < 8; ++v) acc[i][j][v] = 0.0f;

    for (int k0 = 0; k0 < Kdim; k0 += 32) {
        if (k0 + 256 < Kdim) {  // near-cache prefetch of a future K-step
            __builtin_prefetch(pA + fragOff + 256, 0, 3);
            __builtin_prefetch(pW + fragOff + 256, 0, 3);
        }
        v16bf a[4], b[4];
#pragma unroll
        for (int t = 0; t < 4; ++t) {
            const bf16* p = pA + fragOff + t * rowStride16;
            a[t] = combine_frag(*reinterpret_cast<const v8bf*>(p),
                                *reinterpret_cast<const v8bf*>(p + 16));
        }
#pragma unroll
        for (int t = 0; t < 4; ++t) {
            const bf16* p = pW + fragOff + t * rowStride16;
            b[t] = combine_frag(*reinterpret_cast<const v8bf*>(p),
                                *reinterpret_cast<const v8bf*>(p + 16));
        }
#pragma unroll
        for (int i = 0; i < 4; ++i)
#pragma unroll
            for (int j = 0; j < 4; ++j)
                acc[i][j] = __builtin_amdgcn_wmma_f32_16x16x32_bf16(
                    false, a[i], false, b[j], (short)0, acc[i][j], false, false);

        pA += 32;
        pW += 32;
    }

    // Epilogue: C/D layout (ISA 7.12.2): VGPR v -> M = v + (lane>=16 ? 8 : 0),
    // N = lane & 15.
    const int colL   = lane & 15;
    const int rowOff = (lane >> 4) << 3;
#pragma unroll
    for (int i = 0; i < 4; ++i)
#pragma unroll
        for (int j = 0; j < 4; ++j) {
            const int col = n0 + j * 16 + colL;
            const float bv = bias[col];
#pragma unroll
            for (int v = 0; v < 8; ++v) {
                const int row = m0 + i * 16 + rowOff + v;
                D[(size_t)row * TEXT_DIM + col] = (bf16)(acc[i][j][v] + bv);
            }
        }
}

// ---------------------------------------------------------------------------
// Phase 3: fused per-row attention (8 heads x 8 heads over head_dim=128)
//          + residual + LayerNorm. One wave32 per batch row.
// ---------------------------------------------------------------------------
__global__ __launch_bounds__(256)
void attn_ln_kernel(const bf16* __restrict__ Q, const bf16* __restrict__ K,
                    const bf16* __restrict__ V, const float* __restrict__ text,
                    const float* __restrict__ gamma, const float* __restrict__ beta,
                    float* __restrict__ out) {
    const int lane = threadIdx.x & 31;
    const int row  = blockIdx.x * 8 + (threadIdx.x >> 5);
    const size_t base = (size_t)row * TEXT_DIM;
    const int dOff = lane * 4;

    float kreg[NUM_HEADS][4], vreg[NUM_HEADS][4];
#pragma unroll
    for (int h = 0; h < NUM_HEADS; ++h) {
        v4bf kk = *reinterpret_cast<const v4bf*>(K + base + h * HEAD_DIM + dOff);
        v4bf vv = *reinterpret_cast<const v4bf*>(V + base + h * HEAD_DIM + dOff);
#pragma unroll
        for (int j = 0; j < 4; ++j) { kreg[h][j] = (float)kk[j]; vreg[h][j] = (float)vv[j]; }
    }

    float att[NUM_HEADS][4];
#pragma unroll
    for (int h = 0; h < NUM_HEADS; ++h) {
        v4bf qq = *reinterpret_cast<const v4bf*>(Q + base + h * HEAD_DIM + dOff);
        float q[4];
#pragma unroll
        for (int j = 0; j < 4; ++j) q[j] = (float)qq[j];

        float sc[NUM_HEADS];
#pragma unroll
        for (int g = 0; g < NUM_HEADS; ++g) {
            float p = q[0] * kreg[g][0] + q[1] * kreg[g][1]
                    + q[2] * kreg[g][2] + q[3] * kreg[g][3];
#pragma unroll
            for (int off = 16; off > 0; off >>= 1) p += __shfl_xor(p, off, 32);
            sc[g] = p * 0.088388347648318447f;   // 1/sqrt(128)
        }
        float mx = sc[0];
#pragma unroll
        for (int g = 1; g < NUM_HEADS; ++g) mx = fmaxf(mx, sc[g]);
        float s = 0.0f;
#pragma unroll
        for (int g = 0; g < NUM_HEADS; ++g) { sc[g] = __expf(sc[g] - mx); s += sc[g]; }
        const float inv = 1.0f / s;
#pragma unroll
        for (int j = 0; j < 4; ++j) {
            float a = 0.0f;
#pragma unroll
            for (int g = 0; g < NUM_HEADS; ++g) a += sc[g] * vreg[g][j];
            att[h][j] = a * inv;
        }
    }

    // residual + LayerNorm over all 1024 columns
    float sum = 0.0f, ss = 0.0f;
#pragma unroll
    for (int h = 0; h < NUM_HEADS; ++h) {
        float4 t = *reinterpret_cast<const float4*>(text + base + h * HEAD_DIM + dOff);
        att[h][0] += t.x; att[h][1] += t.y; att[h][2] += t.z; att[h][3] += t.w;
#pragma unroll
        for (int j = 0; j < 4; ++j) { sum += att[h][j]; ss += att[h][j] * att[h][j]; }
    }
#pragma unroll
    for (int off = 16; off > 0; off >>= 1) {
        sum += __shfl_xor(sum, off, 32);
        ss  += __shfl_xor(ss,  off, 32);
    }
    const float mean = sum * (1.0f / TEXT_DIM);
    const float var  = ss * (1.0f / TEXT_DIM) - mean * mean;
    const float rstd = rsqrtf(var + 1e-5f);

#pragma unroll
    for (int h = 0; h < NUM_HEADS; ++h) {
        const int c0 = h * HEAD_DIM + dOff;
        float4 g4 = *reinterpret_cast<const float4*>(gamma + c0);
        float4 b4 = *reinterpret_cast<const float4*>(beta + c0);
        float4 o;
        o.x = (att[h][0] - mean) * rstd * g4.x + b4.x;
        o.y = (att[h][1] - mean) * rstd * g4.y + b4.y;
        o.z = (att[h][2] - mean) * rstd * g4.z + b4.z;
        o.w = (att[h][3] - mean) * rstd * g4.w + b4.w;
        *reinterpret_cast<float4*>(out + base + c0) = o;
    }
}

// ---------------------------------------------------------------------------
// Host orchestration
// ---------------------------------------------------------------------------
static inline void launch_cvt(const float* src, bf16* dst, long n, hipStream_t stream) {
    int blocks = (int)((n / 4 + 255) / 256);
    f32_to_bf16_kernel<<<blocks, 256, 0, stream>>>(src, dst, (int)n);
}

extern "C" void kernel_launch(void* const* d_in, const int* in_sizes, int n_in,
                              void* d_out, int out_size, void* d_ws, size_t ws_size,
                              hipStream_t stream) {
    (void)in_sizes; (void)n_in; (void)out_size; (void)ws_size;

    const float* text  = (const float*)d_in[0];
    const float* image = (const float*)d_in[1];
    const float* Wq    = (const float*)d_in[2];
    const float* bq    = (const float*)d_in[3];
    const float* Wk    = (const float*)d_in[4];
    const float* bk    = (const float*)d_in[5];
    const float* Wv    = (const float*)d_in[6];
    const float* bv    = (const float*)d_in[7];
    const float* gamma = (const float*)d_in[8];
    const float* beta  = (const float*)d_in[9];
    float* out = (float*)d_out;

    // workspace layout (all bf16), ~202 MB total
    char* p = (char*)d_ws;
    bf16* textb = (bf16*)p; p += (size_t)BATCH * TEXT_DIM  * sizeof(bf16);
    bf16* imgb  = (bf16*)p; p += (size_t)BATCH * IMAGE_DIM * sizeof(bf16);
    bf16* wqb   = (bf16*)p; p += (size_t)TEXT_DIM * TEXT_DIM  * sizeof(bf16);
    bf16* wkb   = (bf16*)p; p += (size_t)TEXT_DIM * IMAGE_DIM * sizeof(bf16);
    bf16* wvb   = (bf16*)p; p += (size_t)TEXT_DIM * IMAGE_DIM * sizeof(bf16);
    bf16* Qb    = (bf16*)p; p += (size_t)BATCH * TEXT_DIM * sizeof(bf16);
    bf16* Kb    = (bf16*)p; p += (size_t)BATCH * TEXT_DIM * sizeof(bf16);
    bf16* Vb    = (bf16*)p; p += (size_t)BATCH * TEXT_DIM * sizeof(bf16);

    // phase 1: convert to bf16
    launch_cvt(text,  textb, (long)BATCH * TEXT_DIM,      stream);
    launch_cvt(image, imgb,  (long)BATCH * IMAGE_DIM,     stream);
    launch_cvt(Wq,    wqb,   (long)TEXT_DIM * TEXT_DIM,   stream);
    launch_cvt(Wk,    wkb,   (long)TEXT_DIM * IMAGE_DIM,  stream);
    launch_cvt(Wv,    wvb,   (long)TEXT_DIM * IMAGE_DIM,  stream);

    // phase 2: WMMA GEMMs  (grid: N/128 x M/256)
    dim3 ggrid(TEXT_DIM / 128, BATCH / 256);
    gemm_bf16_wmma<<<ggrid, 256, 0, stream>>>(textb, wqb, bq, Qb, TEXT_DIM);
    gemm_bf16_wmma<<<ggrid, 256, 0, stream>>>(imgb,  wkb, bk, Kb, IMAGE_DIM);
    gemm_bf16_wmma<<<ggrid, 256, 0, stream>>>(imgb,  wvb, bv, Vb, IMAGE_DIM);

    // phase 3: fused attention + residual + LayerNorm (1 wave per row)
    attn_ln_kernel<<<BATCH / 8, 256, 0, stream>>>(Qb, Kb, Vb, text, gamma, beta, out);
}
